// GraphCastEncoder_77068893159638
// MI455X (gfx1250) — matrix operand order repaired and checked by reading, not compiled, not run on hardware
//
#include <hip/hip_runtime.h>
#include <hip/hip_bf16.h>

typedef __bf16 bf16_t;
typedef __attribute__((ext_vector_type(16))) __bf16 v16bf;
typedef __attribute__((ext_vector_type(8)))  __bf16 v8bf;
typedef __attribute__((ext_vector_type(8)))  float  v8f;

enum InMode  { IN_DENSE_F32 = 0, IN_DENSE_BF16 = 1, IN_EDGE = 2, IN_NODE = 3 };
enum OutMode { OUT_BF16 = 0, OUT_F32 = 1, OUT_RESID = 2, OUT_EDGE = 3 };

__device__ __forceinline__ float silu_f(float x) {
    // x * sigmoid(x) with fast v_rcp_f32 (bf16 matmul precision already accepted)
    return x * __builtin_amdgcn_rcpf(1.0f + __expf(-x));
}

// 8 consecutive fp32 -> v8bf (two b128 loads + packed converts)
__device__ __forceinline__ v8bf cvt8_f32(const float* __restrict__ p) {
    const float4 f0 = *(const float4*)(p);
    const float4 f1 = *(const float4*)(p + 4);
    v8bf r;
    r[0] = (bf16_t)f0.x; r[1] = (bf16_t)f0.y; r[2] = (bf16_t)f0.z; r[3] = (bf16_t)f0.w;
    r[4] = (bf16_t)f1.x; r[5] = (bf16_t)f1.y; r[6] = (bf16_t)f1.z; r[7] = (bf16_t)f1.w;
    return r;
}

// ---------------------------------------------------------------------------
// Weight pre-swizzle: fp32 W[K x 64] -> bf16 fragments in per-lane B layout.
// Fragment f = (ks*4 + nt)*32 + lane holds 16 contiguous bf16:
//   element i = W[ks*32 + (i<8 ? (lane>>4)*8 + i : 16 + (lane>>4)*8 + i-8)][nt*16 + (lane&15)]  (0 if k >= K)
// Main kernels then read each B fragment as one 32-byte v16bf load.
// ---------------------------------------------------------------------------
__global__ __launch_bounds__(256)
void gc_swz_kernel(const float* __restrict__ w, int K,
                   bf16_t* __restrict__ outw, int nelem)
{
    const int e = blockIdx.x * 256 + threadIdx.x;
    if (e >= nelem) return;
    const int i    = e & 15;
    const int lane = (e >> 4) & 31;
    const int nt   = (e >> 9) & 3;
    const int ks   = e >> 11;
    const int k    = ks * 32 + ((i < 8) ? ((lane >> 4) * 8 + i) : (16 + (lane >> 4) * 8 + (i - 8)));
    const int n    = nt * 16 + (lane & 15);
    outw[e] = (k < K) ? (bf16_t)w[(size_t)k * 64 + n] : (bf16_t)0.0f;
}

// ---------------------------------------------------------------------------
// One wave computes a 16-row tile of:  LN(SiLU(A @ W1 + b1) @ W2 + b2)
// A is [16, DIN]: dense fp32 features, dense bf16 activations, or gather-concat
// of bf16 activations (+ fp32 agg for the node MLP). bf16 WMMA operands,
// fp32 accumulate, fp32 SiLU/LayerNorm/residual math.
// ---------------------------------------------------------------------------
template<int DIN, int IM, int OM>
__global__ __launch_bounds__(256)
void gc_mlp_kernel(const float*  __restrict__ xf,     // IN_DENSE_F32 input
                   const bf16_t* __restrict__ g0,     // seg0 bf16 (grid_e/mesh_e)
                   const bf16_t* __restrict__ g1,     // seg1 bf16 (mesh_e, edge)
                   const float*  __restrict__ f1,     // seg1 fp32 (agg, node)
                   const bf16_t* __restrict__ g2,     // seg2 bf16 (g2m_e, edge)
                   const int*    __restrict__ src_idx,
                   const int*    __restrict__ dst_idx,
                   const bf16_t* __restrict__ w1s,  const float* __restrict__ b1,
                   const bf16_t* __restrict__ w2s,  const float* __restrict__ b2,
                   const float* __restrict__ gamma, const float* __restrict__ beta,
                   const bf16_t* __restrict__ resid16,
                   float*  __restrict__ outf,
                   bf16_t* __restrict__ out16,
                   float*  __restrict__ aggp,
                   int N)
{
    // per-wave H tile, stride 68 floats (272B: 16B-aligned rows, bank-spread)
    __shared__ __align__(16) float sH[8 * 16 * 68];

    const int lane  = threadIdx.x & 31;
    const int wid   = threadIdx.x >> 5;      // wave id in block (8 waves)
    const int rlane = lane & 15;
    const int hi    = lane >> 4;             // half-wave select
    const int ko    = hi * 8;                // K octet base for A fragments
    const int tile  = blockIdx.x * 8 + wid;

    const int row  = tile * 16 + rlane;      // A-matrix row handled by this lane
    const int rowc = (row < N) ? row : (N - 1);

    // Per-lane input row pointers (constant across all K-steps).
    const float*  pF  = nullptr;
    const bf16_t* pB0 = nullptr;
    const bf16_t* pB1 = nullptr;
    const bf16_t* pB2 = nullptr;
    if constexpr (IM == IN_DENSE_F32) {
        pF = xf + (size_t)rowc * DIN;
    } else if constexpr (IM == IN_DENSE_BF16) {
        pB0 = g0 + (size_t)rowc * 64;
    } else if constexpr (IM == IN_NODE) {
        pB0 = g0 + (size_t)rowc * 64;                 // mesh_e (bf16)
        pF  = f1 + (size_t)rowc * 64;                 // agg    (fp32)
    } else { // IN_EDGE: concat(sender, receiver, g2m_e), all bf16
        pB0 = g0 + (size_t)src_idx[rowc] * 64;        // grid_e[src]
        pB1 = g1 + (size_t)dst_idx[rowc] * 64;        // mesh_e[dst]
        pB2 = g2 + (size_t)rowc * 64;                 // g2m_e
    }

    // 8-element A run at logical column k0 (k0 multiple of 8; runs never cross
    // a 64-aligned concat-segment boundary).
    auto run8 = [&](int k0) -> v8bf {
        if constexpr (IM == IN_DENSE_F32) {
            return cvt8_f32(pF + k0);
        } else if constexpr (IM == IN_DENSE_BF16) {
            return *(const v8bf*)(pB0 + k0);
        } else if constexpr (IM == IN_NODE) {
            return (k0 < 64) ? *(const v8bf*)(pB0 + k0) : cvt8_f32(pF + (k0 - 64));
        } else {
            const bf16_t* p = (k0 < 64) ? (pB0 + k0)
                            : ((k0 < 128) ? (pB1 + (k0 - 64)) : (pB2 + (k0 - 128)));
            return *(const v8bf*)p;
        }
    };

    v8f acc[4];
    #pragma unroll
    for (int nt = 0; nt < 4; ++nt)
        #pragma unroll
        for (int e = 0; e < 8; ++e) acc[nt][e] = 0.0f;

    // ---------------- Stage 1: H = SiLU(A @ W1 + b1) ----------------
    constexpr int KS1 = (DIN + 31) / 32;
    #pragma unroll
    for (int ks = 0; ks < KS1; ++ks) {
        // A fragment: 16-bit A 16x32 layout (lane=row; hi half selects K octets)
        v16bf afrag;
        if constexpr (DIN % 32 == 0) {
            const v8bf lo = run8(ks * 32 + ko);
            const v8bf hiv = run8(ks * 32 + 16 + ko);
            afrag = __builtin_shufflevector(lo, hiv,
                0, 1, 2, 3, 4, 5, 6, 7, 8, 9, 10, 11, 12, 13, 14, 15);
        } else {
            // only the tiny fp32 embeds (DIN = 3, 4) take this path
            #pragma unroll
            for (int i = 0; i < 16; ++i) {
                const int k = ks * 32 + ((i < 8) ? (ko + i) : (16 + ko + (i - 8)));
                float v = 0.0f;
                if (k < DIN) v = pF[k];
                afrag[i] = (bf16_t)v;
            }
        }
        #pragma unroll
        for (int nt = 0; nt < 4; ++nt) {
            const v16bf bfrag =
                *(const v16bf*)(w1s + (((size_t)ks * 4 + nt) * 32 + lane) * 16);
            acc[nt] = __builtin_amdgcn_wmma_f32_16x16x32_bf16(
                false, afrag, false, bfrag, (short)0, acc[nt], false, false);
        }
    }

    // bias + SiLU, spill H tile to LDS (re-layout D -> A)
    float* hrow = &sH[wid * 16 * 68];
    #pragma unroll
    for (int nt = 0; nt < 4; ++nt) {
        const int n  = nt * 16 + rlane;
        const float bb = b1[n];
        #pragma unroll
        for (int r = 0; r < 8; ++r) {
            hrow[(hi * 8 + r) * 68 + n] = silu_f(acc[nt][r] + bb);
        }
    }
    __syncthreads();

    // ---------------- Stage 2: Y = H @ W2 + b2 ----------------
    #pragma unroll
    for (int nt = 0; nt < 4; ++nt)
        #pragma unroll
        for (int e = 0; e < 8; ++e) acc[nt][e] = 0.0f;

    #pragma unroll
    for (int ks = 0; ks < 2; ++ks) {
        const float* hp0 = &hrow[rlane * 68 + ks * 32 + ko];
        const float4 h0 = *(const float4*)(hp0);
        const float4 h1 = *(const float4*)(hp0 + 4);
        const float4 h2 = *(const float4*)(hp0 + 16);
        const float4 h3 = *(const float4*)(hp0 + 20);
        v16bf afrag;
        afrag[0]  = (bf16_t)h0.x; afrag[1]  = (bf16_t)h0.y;
        afrag[2]  = (bf16_t)h0.z; afrag[3]  = (bf16_t)h0.w;
        afrag[4]  = (bf16_t)h1.x; afrag[5]  = (bf16_t)h1.y;
        afrag[6]  = (bf16_t)h1.z; afrag[7]  = (bf16_t)h1.w;
        afrag[8]  = (bf16_t)h2.x; afrag[9]  = (bf16_t)h2.y;
        afrag[10] = (bf16_t)h2.z; afrag[11] = (bf16_t)h2.w;
        afrag[12] = (bf16_t)h3.x; afrag[13] = (bf16_t)h3.y;
        afrag[14] = (bf16_t)h3.z; afrag[15] = (bf16_t)h3.w;
        #pragma unroll
        for (int nt = 0; nt < 4; ++nt) {
            const v16bf bfrag =
                *(const v16bf*)(w2s + (((size_t)ks * 4 + nt) * 32 + lane) * 16);
            acc[nt] = __builtin_amdgcn_wmma_f32_16x16x32_bf16(
                false, afrag, false, bfrag, (short)0, acc[nt], false, false);
        }
    }

    // ---------------- LayerNorm (fp32, per row of 64) ----------------
    float b2v[4], gv[4], bev[4];
    #pragma unroll
    for (int nt = 0; nt < 4; ++nt) {
        const int n = nt * 16 + rlane;
        b2v[nt] = b2[n]; gv[nt] = gamma[n]; bev[nt] = beta[n];
    }
    #pragma unroll
    for (int nt = 0; nt < 4; ++nt)
        #pragma unroll
        for (int r = 0; r < 8; ++r) acc[nt][r] += b2v[nt];

    float mean[8], rstd[8];
    #pragma unroll
    for (int r = 0; r < 8; ++r) {
        float s = acc[0][r] + acc[1][r] + acc[2][r] + acc[3][r];
        float q = acc[0][r] * acc[0][r] + acc[1][r] * acc[1][r]
                + acc[2][r] * acc[2][r] + acc[3][r] * acc[3][r];
        #pragma unroll
        for (int m = 1; m < 16; m <<= 1) {       // reduce within 16-lane half-wave
            s += __shfl_xor(s, m, 32);
            q += __shfl_xor(q, m, 32);
        }
        const float mu  = s * (1.0f / 64.0f);
        const float var = q * (1.0f / 64.0f) - mu * mu;
        mean[r] = mu;
        rstd[r] = __builtin_amdgcn_rsqf(var + 1e-5f);   // fast v_rsq_f32
    }

    // ---------------- Epilogue: affine, residual / scatter, store ----------------
    #pragma unroll
    for (int r = 0; r < 8; ++r) {
        const int  orow = tile * 16 + hi * 8 + r;   // row owned by this (lane,r)
        const bool ok   = orow < N;
        int dsti = 0;
        if constexpr (OM == OUT_EDGE) { if (ok) dsti = dst_idx[orow]; }
        #pragma unroll
        for (int nt = 0; nt < 4; ++nt) {
            const int   n = nt * 16 + rlane;
            const float y = (acc[nt][r] - mean[r]) * rstd[r] * gv[nt] + bev[nt];
            if constexpr (OM == OUT_BF16) {
                if (ok) out16[(size_t)orow * 64 + n] = (bf16_t)y;
            } else if constexpr (OM == OUT_F32) {
                if (ok) outf[(size_t)orow * 64 + n] = y;
            } else if constexpr (OM == OUT_RESID) {
                if (ok) outf[(size_t)orow * 64 + n] =
                    y + (float)resid16[(size_t)orow * 64 + n];
            } else { // OUT_EDGE: scatter e_upd into agg, g2m_out = g2m_e + e_upd
                if (ok) {
                    atomicAdd(&aggp[(size_t)dsti * 64 + n], y);
                    outf[(size_t)orow * 64 + n] =
                        y + (float)resid16[(size_t)orow * 64 + n];
                }
            }
        }
    }
}

extern "C" void kernel_launch(void* const* d_in, const int* in_sizes, int n_in,
                              void* d_out, int out_size, void* d_ws, size_t ws_size,
                              hipStream_t stream)
{
    (void)in_sizes; (void)n_in; (void)out_size; (void)ws_size;

    constexpr int NG = 1038240, NM = 40962, EG = 1600000, EM = 245772;

    const float* grid_x   = (const float*)d_in[0];
    const float* mesh_x   = (const float*)d_in[1];
    const float* mm_attr  = (const float*)d_in[2];
    const float* g2m_attr = (const float*)d_in[3];
    const int*   src_idx  = (const int*)d_in[4];
    const int*   dst_idx  = (const int*)d_in[5];

    // params in setup_inputs() insertion order:
    // 0 grid_embed, 1 mesh_embed, 2 mesh_edge_embed, 3 edge_embed_g2m,
    // 4 grid_update, 5 edge_mlp, 6 node_mlp ; each {w1,b1,w2,b2,gamma,beta}
    auto W = [&](int m, int j) { return (const float*)d_in[6 + m * 6 + j]; };

    // ---- workspace layout (base is 256B aligned; all regions 16B aligned) ----
    bf16_t* grid_e16 = (bf16_t*)d_ws;                       // [NG,64] bf16
    bf16_t* mesh_e16 = grid_e16 + (size_t)NG * 64;          // [NM,64] bf16
    bf16_t* g2m_e16  = mesh_e16 + (size_t)NM * 64;          // [EG,64] bf16
    float*  agg      = (float*)(g2m_e16 + (size_t)EG * 64); // [NM,64] fp32

    constexpr int dins[7] = {32, 3, 4, 4, 64, 192, 128};
    bf16_t* cur = (bf16_t*)(agg + (size_t)NM * 64);         // swizzled weights
    bf16_t* w1s[7]; bf16_t* w2s[7];
    for (int m = 0; m < 7; ++m) {
        const int ks = (dins[m] + 31) / 32;
        w1s[m] = cur; cur += (size_t)ks * 2048;             // ks*4*32*16 bf16
        w2s[m] = cur; cur += (size_t)2 * 2048;              // K=64 -> 2 K-steps
    }

    float* out      = (float*)d_out;
    float* grid_out = out;
    float* mesh_out = grid_out + (size_t)NG * 64;
    float* mm_out   = mesh_out + (size_t)NM * 64;
    float* g2m_out  = mm_out   + (size_t)EM * 64;

    hipMemsetAsync(agg, 0, (size_t)NM * 64 * sizeof(float), stream);

    // one-time (per call) weight swizzle to bf16 fragment layout
    for (int m = 0; m < 7; ++m) {
        const int ks = (dins[m] + 31) / 32;
        const int n1 = ks * 2048;
        gc_swz_kernel<<<(n1 + 255) / 256, 256, 0, stream>>>(W(m, 0), dins[m], w1s[m], n1);
        gc_swz_kernel<<<(2 * 2048 + 255) / 256, 256, 0, stream>>>(W(m, 2), 64, w2s[m], 2 * 2048);
    }

    auto blocks = [](int n) { return ((n + 15) / 16 + 7) / 8; };
    dim3 T(256);

    // grid_e = MLP(grid_x)               -> bf16 ws
    gc_mlp_kernel<32, IN_DENSE_F32, OUT_BF16><<<blocks(NG), T, 0, stream>>>(
        grid_x, nullptr, nullptr, nullptr, nullptr, nullptr, nullptr,
        w1s[0], W(0,1), w2s[0], W(0,3), W(0,4), W(0,5),
        nullptr, nullptr, grid_e16, nullptr, NG);

    // mesh_e = MLP(mesh_x)               -> bf16 ws
    gc_mlp_kernel<3, IN_DENSE_F32, OUT_BF16><<<blocks(NM), T, 0, stream>>>(
        mesh_x, nullptr, nullptr, nullptr, nullptr, nullptr, nullptr,
        w1s[1], W(1,1), w2s[1], W(1,3), W(1,4), W(1,5),
        nullptr, nullptr, mesh_e16, nullptr, NM);

    // mm_e = MLP(mesh_edge_attr)         -> final fp32 output
    gc_mlp_kernel<4, IN_DENSE_F32, OUT_F32><<<blocks(EM), T, 0, stream>>>(
        mm_attr, nullptr, nullptr, nullptr, nullptr, nullptr, nullptr,
        w1s[2], W(2,1), w2s[2], W(2,3), W(2,4), W(2,5),
        nullptr, mm_out, nullptr, nullptr, EM);

    // g2m_e = MLP(g2m_edge_attr)         -> bf16 ws
    gc_mlp_kernel<4, IN_DENSE_F32, OUT_BF16><<<blocks(EG), T, 0, stream>>>(
        g2m_attr, nullptr, nullptr, nullptr, nullptr, nullptr, nullptr,
        w1s[3], W(3,1), w2s[3], W(3,3), W(3,4), W(3,5),
        nullptr, nullptr, g2m_e16, nullptr, EG);

    // edge_mlp over concat(grid_e[src], mesh_e[dst], g2m_e), all bf16;
    // scatter-adds e_upd into fp32 agg, writes g2m_out = g2m_e + e_upd (fp32).
    gc_mlp_kernel<192, IN_EDGE, OUT_EDGE><<<blocks(EG), T, 0, stream>>>(
        nullptr, grid_e16, mesh_e16, nullptr, g2m_e16, src_idx, dst_idx,
        w1s[5], W(5,1), w2s[5], W(5,3), W(5,4), W(5,5),
        g2m_e16, g2m_out, nullptr, agg, EG);

    // mesh_out = mesh_e + MLP(concat(mesh_e bf16, agg fp32))
    gc_mlp_kernel<128, IN_NODE, OUT_RESID><<<blocks(NM), T, 0, stream>>>(
        nullptr, mesh_e16, nullptr, agg, nullptr, nullptr, dst_idx,
        w1s[6], W(6,1), w2s[6], W(6,3), W(6,4), W(6,5),
        mesh_e16, mesh_out, nullptr, nullptr, NM);

    // grid_out = grid_e + MLP(grid_e)    (bf16 in, fp32 out)
    gc_mlp_kernel<64, IN_DENSE_BF16, OUT_RESID><<<blocks(NG), T, 0, stream>>>(
        nullptr, grid_e16, nullptr, nullptr, nullptr, nullptr, nullptr,
        w1s[4], W(4,1), w2s[4], W(4,3), W(4,4), W(4,5),
        grid_e16, grid_out, nullptr, nullptr, NG);
}
